// GraphTransformerNet_40261023432798
// MI455X (gfx1250) — compile-verified, matrix operand order checked by reading, not run on hardware
//
#include <hip/hip_runtime.h>
#include <hip/hip_bf16.h>
#include <math.h>

// ---------------------------------------------------------------------------
// GCN (3-layer GCNConv + sum-pool) for MI455X / gfx1250.
//   Layer: h = x @ W (WMMA f16->f32), agg = CSR gather-sum of norm_e*h[src],
//          out = agg + norm_self*h + b. Final: column-sum / sqrt(N).
// CSR is built once per launch (histogram -> scan -> reorder) and reused for
// all 3 layers, eliminating ~230M fp32 scatter atomics.
// ---------------------------------------------------------------------------

typedef __attribute__((ext_vector_type(16))) _Float16 v16h;
typedef __attribute__((ext_vector_type(8)))  float    v8f;

constexpr int DF = 128;          // feature dim
constexpr int WT_STRIDE = 132;   // padded LDS stride (halfs): 264B, 8B-aligned,
                                 // breaks the 64-way bank conflict of stride 128

// ---------------- small utility kernels ----------------
__global__ void zero_i32(int* p, int n) {
  int i = blockIdx.x * blockDim.x + threadIdx.x;
  if (i < n) p[i] = 0;
}
__global__ void zero_f32(float* p, int n) {
  int i = blockIdx.x * blockDim.x + threadIdx.x;
  if (i < n) p[i] = 0.0f;
}

// histogram of destination nodes -> degree (without self loop)
__global__ void hist_kernel(const int* __restrict__ dst, int* __restrict__ deg, int e) {
  int i = blockIdx.x * blockDim.x + threadIdx.x;
  if (i < e) atomicAdd(&deg[dst[i]], 1);
}

// dinv = rsqrt(deg + 1)   (self loop included)
__global__ void dinv_kernel(const int* __restrict__ deg, float* __restrict__ dinv, int n) {
  int i = blockIdx.x * blockDim.x + threadIdx.x;
  if (i < n) dinv[i] = rsqrtf((float)deg[i] + 1.0f);
}

// ---------------- exclusive scan (3 phases) over degree -> row_ptr ----------
__global__ void scan1_kernel(const int* __restrict__ deg, int* __restrict__ tmp,
                             int* __restrict__ bsum, int n) {
  __shared__ int s[256];
  int tid = threadIdx.x;
  int i = blockIdx.x * 256 + tid;
  s[tid] = (i < n) ? deg[i] : 0;
  __syncthreads();
  for (int off = 1; off < 256; off <<= 1) {
    int t = (tid >= off) ? s[tid - off] : 0;
    __syncthreads();
    s[tid] += t;
    __syncthreads();
  }
  if (i < n) tmp[i] = s[tid];
  if (tid == 255) bsum[blockIdx.x] = s[255];
}

__global__ void scan2_kernel(const int* __restrict__ bsum, int* __restrict__ bscan, int nblk) {
  __shared__ int s[1024];
  int tid = threadIdx.x;
  s[tid] = (tid < nblk) ? bsum[tid] : 0;
  __syncthreads();
  for (int off = 1; off < 1024; off <<= 1) {
    int t = (tid >= off) ? s[tid - off] : 0;
    __syncthreads();
    s[tid] += t;
    __syncthreads();
  }
  if (tid < nblk) bscan[tid] = s[tid];
}

__global__ void scan3_kernel(const int* __restrict__ tmp, const int* __restrict__ bscan,
                             int* __restrict__ row_ptr, int n) {
  int i = blockIdx.x * 256 + threadIdx.x;
  if (i < n) {
    int add = (blockIdx.x > 0) ? bscan[blockIdx.x - 1] : 0;
    row_ptr[i + 1] = tmp[i] + add;
  }
  if (i == 0) row_ptr[0] = 0;
}

// reorder edges into CSR; fuse edge-normalization weight computation
__global__ void csr_fill_kernel(const int* __restrict__ src, const int* __restrict__ dst,
                                const int* __restrict__ row_ptr, int* __restrict__ cursor,
                                const float* __restrict__ dinv, int* __restrict__ esrc,
                                float* __restrict__ ew, int e) {
  int i = blockIdx.x * blockDim.x + threadIdx.x;
  if (i < e) {
    int d = dst[i];
    int s = src[i];
    int pos = row_ptr[d] + atomicAdd(&cursor[d], 1);
    esrc[pos] = s;
    ew[pos] = dinv[s] * dinv[d];
  }
}

// fp32 weights -> f16 (all L layers at once)
__global__ void wconv_kernel(const float* __restrict__ W, _Float16* __restrict__ Wh, int n) {
  int i = blockIdx.x * blockDim.x + threadIdx.x;
  if (i < n) Wh[i] = (_Float16)W[i];
}

// ---------------- WMMA GEMM: h[N,128] = x[N,128] @ W[128,128] ---------------
// 256 threads = 8 waves; each wave owns a 16-row strip and all 8 col tiles.
// W staged transposed in LDS (Wt[col][k], padded stride) so B fragments are
// two contiguous ds_load_b64 per tile; A fragments are contiguous b128 loads
// of the x row converted f32->f16.
__global__ __launch_bounds__(256) void gemm_kernel(const float* __restrict__ x,
                                                   const _Float16* __restrict__ Wh,
                                                   float* __restrict__ h, int nrows) {
  __shared__ _Float16 Wt[DF * WT_STRIDE];
  int tid = threadIdx.x;
  // cooperative transpose fill: coalesced global reads; padded LDS stride
  for (int idx = tid; idx < DF * DF; idx += 256) {
    int k = idx >> 7;
    int c = idx & 127;
    Wt[c * WT_STRIDE + k] = Wh[idx];
  }
  __syncthreads();

  int wave = tid >> 5;
  int lane = tid & 31;
  int halfSel = (lane < 16) ? 0 : 8;   // K-half selector per WMMA A/B layout
  int mrow = lane & 15;                // row (A) / col (B) within tile
  int r0 = (blockIdx.x * 8 + wave) * 16;
  int rowA = r0 + mrow;
  int rowC = (rowA < nrows) ? rowA : (nrows - 1);  // clamp OOB reads
  const float* xrow = x + (size_t)rowC * DF;

  v8f acc[8];
  v8f zero = {0.f, 0.f, 0.f, 0.f, 0.f, 0.f, 0.f, 0.f};
#pragma unroll
  for (int t = 0; t < 8; t++) acc[t] = zero;

#pragma unroll
  for (int k0 = 0; k0 < DF; k0 += 32) {
    int kb = k0 + halfSel;
    v16h a;
#pragma unroll
    for (int i = 0; i < 8; i++) a[i] = (_Float16)xrow[kb + i];
#pragma unroll
    for (int i = 0; i < 8; i++) a[8 + i] = (_Float16)xrow[kb + 16 + i];

#pragma unroll
    for (int nt = 0; nt < 8; nt++) {
      int col = nt * 16 + mrow;
      const _Float16* wt = &Wt[col * WT_STRIDE + kb];
      v16h b;
#pragma unroll
      for (int i = 0; i < 8; i++) b[i] = wt[i];
#pragma unroll
      for (int i = 0; i < 8; i++) b[8 + i] = wt[16 + i];
      acc[nt] = __builtin_amdgcn_wmma_f32_16x16x32_f16(
          false, a, false, b, (short)0, acc[nt], false, false);
    }
  }

  // C/D layout: lane gives col, VGPR j gives row j (+8 for upper half-wave)
#pragma unroll
  for (int nt = 0; nt < 8; nt++) {
#pragma unroll
    for (int j = 0; j < 8; j++) {
      int row = r0 + j + halfSel;
      if (row < nrows) h[(size_t)row * DF + nt * 16 + mrow] = acc[nt][j];
    }
  }
}

// ---------------- CSR aggregation: one wave32 per node ----------------------
// acc = norm_self*h[v] + b + sum_e ew[e]*h[esrc[e]]; lane covers float4 of D.
__global__ __launch_bounds__(256) void agg_kernel(const float* __restrict__ h,
                                                  const int* __restrict__ row_ptr,
                                                  const int* __restrict__ esrc,
                                                  const float* __restrict__ ew,
                                                  const float* __restrict__ dinv,
                                                  const float* __restrict__ bias,
                                                  float* __restrict__ xout, int n) {
  int v = blockIdx.x * 8 + (threadIdx.x >> 5);
  int lane = threadIdx.x & 31;
  if (v >= n) return;

  float di = dinv[v];
  float ns = di * di;
  const float4* hv = (const float4*)(h + (size_t)v * DF);
  const float4* bb = (const float4*)bias;
  float4 hs = hv[lane];
  float4 bv = bb[lane];
  float4 acc;
  acc.x = hs.x * ns + bv.x;
  acc.y = hs.y * ns + bv.y;
  acc.z = hs.z * ns + bv.z;
  acc.w = hs.w * ns + bv.w;

  int beg = row_ptr[v];
  int end = row_ptr[v + 1];
  for (int p = beg; p < end; ++p) {
    int s = esrc[p];     // uniform across wave
    float w = ew[p];     // uniform across wave
    const float4* hr = (const float4*)(h + (size_t)s * DF);
    float4 m = hr[lane];
    acc.x += w * m.x;
    acc.y += w * m.y;
    acc.z += w * m.z;
    acc.w += w * m.w;
  }
  float4* out = (float4*)(xout + (size_t)v * DF);
  out[lane] = acc;
}

// ---------------- final sum-pool / sqrt(N) ----------------------------------
__global__ void reduce_kernel(const float* __restrict__ x, float* __restrict__ out,
                              int n, float scale) {
  int t = threadIdx.x;  // 128 threads = one feature column each
  float s = 0.0f;
  for (int v = blockIdx.x; v < n; v += gridDim.x) s += x[(size_t)v * DF + t];
  atomicAdd(&out[t], s * scale);
}

// ---------------------------------------------------------------------------
extern "C" void kernel_launch(void* const* d_in, const int* in_sizes, int n_in,
                              void* d_out, int out_size, void* d_ws, size_t ws_size,
                              hipStream_t stream) {
  const int*   edge_index = (const int*)d_in[0];
  const float* xfeat      = (const float*)d_in[1];
  const float* Ws         = (const float*)d_in[2];
  const float* bs         = (const float*)d_in[3];

  const int E = in_sizes[0] / 2;
  const int N = in_sizes[1] / DF;
  const int L = in_sizes[2] / (DF * DF);
  const int* src = edge_index;
  const int* dst = edge_index + E;

  // carve scratch out of d_ws (256B-aligned slabs)
  size_t off = 0;
  auto carve = [&](size_t bytes) -> void* {
    void* p = (char*)d_ws + off;
    off += (bytes + 255) & ~(size_t)255;
    return p;
  };
  float*     bufA    = (float*)carve((size_t)N * DF * sizeof(float));   // x ping
  float*     bufB    = (float*)carve((size_t)N * DF * sizeof(float));   // h
  float*     dinv    = (float*)carve((size_t)N * sizeof(float));
  int*       deg     = (int*)carve((size_t)N * sizeof(int));
  int*       cursor  = (int*)carve((size_t)N * sizeof(int));
  int*       row_ptr = (int*)carve((size_t)(N + 1) * sizeof(int));
  int*       tmp     = (int*)carve((size_t)N * sizeof(int));
  int*       bsum    = (int*)carve(1024 * sizeof(int));
  int*       bscan   = (int*)carve(1024 * sizeof(int));
  int*       esrc    = (int*)carve((size_t)E * sizeof(int));
  float*     ew      = (float*)carve((size_t)E * sizeof(float));
  _Float16*  WhAll   = (_Float16*)carve((size_t)L * DF * DF * sizeof(_Float16));
  (void)ws_size;

  const int TB = 256;
  int gN = (N + TB - 1) / TB;
  int gE = (E + TB - 1) / TB;
  int nblk = (N + 255) / 256;  // scan blocks (391 for N=100k; <=1024 supported)

  // ---- graph structure (once per launch, reused for all layers) ----
  zero_i32<<<gN, TB, 0, stream>>>(deg, N);
  hist_kernel<<<gE, TB, 0, stream>>>(dst, deg, E);
  dinv_kernel<<<gN, TB, 0, stream>>>(deg, dinv, N);
  scan1_kernel<<<nblk, 256, 0, stream>>>(deg, tmp, bsum, N);
  scan2_kernel<<<1, 1024, 0, stream>>>(bsum, bscan, nblk);
  scan3_kernel<<<nblk, 256, 0, stream>>>(tmp, bscan, row_ptr, N);
  zero_i32<<<gN, TB, 0, stream>>>(cursor, N);
  csr_fill_kernel<<<gE, TB, 0, stream>>>(src, dst, row_ptr, cursor, dinv, esrc, ew, E);

  // ---- weights -> f16 (all layers) ----
  int nW = L * DF * DF;
  wconv_kernel<<<(nW + TB - 1) / TB, TB, 0, stream>>>(Ws, WhAll, nW);

  // ---- layers ----
  int gGemm = (N + 127) / 128;  // 8 waves/block x 16 rows/wave
  int gAgg  = (N + 7) / 8;      // 8 waves/block, one node per wave
  const float* xin = xfeat;
  for (int l = 0; l < L; ++l) {
    gemm_kernel<<<gGemm, 256, 0, stream>>>(xin, WhAll + (size_t)l * DF * DF, bufB, N);
    agg_kernel<<<gAgg, 256, 0, stream>>>(bufB, row_ptr, esrc, ew, dinv,
                                         bs + (size_t)l * DF, bufA, N);
    xin = bufA;
  }

  // ---- sum-pool / sqrt(N) ----
  zero_f32<<<1, DF, 0, stream>>>((float*)d_out, DF);
  float scale = 1.0f / sqrtf((float)N);
  reduce_kernel<<<512, DF, 0, stream>>>(bufA, (float*)d_out, N, scale);
}
